// Translator_68650757259304
// MI455X (gfx1250) — matrix operand order repaired and checked by reading, not compiled, not run on hardware
//
#include <hip/hip_runtime.h>
#include <hip/hip_bf16.h>

// ---------------------------------------------------------------------------
// Types for CDNA5 WMMA
// ---------------------------------------------------------------------------
typedef __bf16 bf16_t;
typedef __attribute__((ext_vector_type(16))) __bf16 v16bf;
typedef __attribute__((ext_vector_type(8)))  float  v8f;

#define B_ROWS 4096
#define DM     1024
#define DE     1024
#define DR     2048
#define NEXP   8
#define BK     32

// ---------------------------------------------------------------------------
// fp32 -> bf16 elementwise convert
// ---------------------------------------------------------------------------
__global__ void cvt_f32_bf16(const float* __restrict__ in, bf16_t* __restrict__ out, int n) {
    int i = blockIdx.x * blockDim.x + threadIdx.x;
    if (i < n) out[i] = (bf16_t)in[i];
}

// ---------------------------------------------------------------------------
// fp32 [batch][K][N] -> bf16 [batch][N][K]  (transpose+convert, 32x32 LDS tile)
// ---------------------------------------------------------------------------
__global__ __launch_bounds__(256) void tconv_kernel(const float* __restrict__ in,
                                                    bf16_t* __restrict__ out,
                                                    int K, int N) {
    __shared__ float tile[32][33];
    const size_t e = blockIdx.z;
    const float* src = in + e * (size_t)K * N;
    bf16_t* dst = out + e * (size_t)N * K;
    int n0 = blockIdx.x * 32, k0 = blockIdx.y * 32;
    int tx = threadIdx.x & 31, ty = threadIdx.x >> 5;      // ty in 0..7
    for (int i = 0; i < 32; i += 8)
        tile[ty + i][tx] = src[(size_t)(k0 + ty + i) * N + (n0 + tx)];
    __syncthreads();
    for (int i = 0; i < 32; i += 8)
        dst[(size_t)(n0 + ty + i) * K + (k0 + tx)] = (bf16_t)tile[tx][ty + i];
}

// ---------------------------------------------------------------------------
// Gate MLP for both MoE branches: softmax(relu(x@gw1+gb1)@gw2+gb2) -> [B,8]
// one block (128 threads) per batch row
// ---------------------------------------------------------------------------
__global__ __launch_bounds__(128) void gate_kernel(const float* __restrict__ x,
        const float* gw1a, const float* gb1a, const float* gw2a, const float* gb2a,
        const float* gw1b, const float* gb1b, const float* gw2b, const float* gb2b,
        float* __restrict__ gateA, float* __restrict__ gateB) {
    __shared__ float xr[DM];
    __shared__ float h[2][64];
    __shared__ float lg[2][8];
    const int b = blockIdx.x, t = threadIdx.x;
    for (int i = t; i < DM; i += 128) xr[i] = x[(size_t)b * DM + i];
    __syncthreads();
    {
        int br = t >> 6, c = t & 63;
        const float* w1 = br ? gw1b : gw1a;
        const float* b1 = br ? gb1b : gb1a;
        float s = b1[c];
        for (int k = 0; k < DM; ++k) s += xr[k] * w1[(size_t)k * 64 + c];
        h[br][c] = fmaxf(s, 0.f);
    }
    __syncthreads();
    if (t < 16) {
        int br = t >> 3, e = t & 7;
        const float* w2 = br ? gw2b : gw2a;
        const float* b2 = br ? gb2b : gb2a;
        float s = b2[e];
        for (int k = 0; k < 64; ++k) s += h[br][k] * w2[k * 8 + e];
        lg[br][e] = s;
    }
    __syncthreads();
    if (t < 2) {
        float mx = -1e30f;
        for (int e = 0; e < 8; ++e) mx = fmaxf(mx, lg[t][e]);
        float ex[8], sum = 0.f;
        for (int e = 0; e < 8; ++e) { ex[e] = __expf(lg[t][e] - mx); sum += ex[e]; }
        float inv = 1.f / sum;
        float* g = t ? gateB : gateA;
        for (int e = 0; e < 8; ++e) g[(size_t)b * 8 + e] = ex[e] * inv;
    }
}

// ---------------------------------------------------------------------------
// SE layer: latent = mu * sigmoid(relu(mu@se_w1)@se_w2), output bf16
// ---------------------------------------------------------------------------
__global__ __launch_bounds__(128) void se_kernel(const float* __restrict__ mu,
                                                 const float* __restrict__ w1,
                                                 const float* __restrict__ w2,
                                                 bf16_t* __restrict__ latent) {
    __shared__ float mrow[DE];
    __shared__ float s16[16];
    const int b = blockIdx.x, t = threadIdx.x;
    for (int i = t; i < DE; i += 128) mrow[i] = mu[(size_t)b * DE + i];
    __syncthreads();
    if (t < 16) {
        float s = 0.f;
        for (int k = 0; k < DE; ++k) s += mrow[k] * w1[(size_t)k * 16 + t];
        s16[t] = fmaxf(s, 0.f);
    }
    __syncthreads();
    for (int c = t; c < DE; c += 128) {
        float z = 0.f;
        for (int k = 0; k < 16; ++k) z += s16[k] * w2[k * DE + c];
        latent[(size_t)b * DE + c] = (bf16_t)(mrow[c] * (1.f / (1.f + __expf(-z))));
    }
}

// ---------------------------------------------------------------------------
// Core WMMA tile loop: block = 256 thr (8 waves), block tile 128x128,
// wave tile 32x64 = 2x4 v_wmma_f32_16x16x32_bf16.
// Register-pipelined double-buffered LDS: tile k+1 global loads overlap with
// tile k WMMAs; exactly one s_barrier per K-tile.
// A: bf16 row-major [M][K]. Bt: bf16 [N][K] (pre-transposed weight).
// ---------------------------------------------------------------------------
__device__ __forceinline__ void gemm_core(const bf16_t* __restrict__ A,
                                          const bf16_t* __restrict__ Bt,
                                          int K, int rowBase, int colBase,
                                          bf16_t (*As)[128][48], bf16_t (*Bs)[128][48],
                                          v8f acc[2][4]) {
    const int tid  = threadIdx.x;
    const int lane = tid & 31;
    const int wave = tid >> 5;
    const int wr = wave >> 1, wc = wave & 1;

    // Fixed per-thread staging assignment: rows r and r+64, byte segment seg.
    const int r   = tid >> 2;        // 0..63
    const int seg = (tid & 3) * 8;   // bf16 element offset within the 32-wide tile
    const size_t rowSkip = (size_t)64 * K;
    const bf16_t* gA = A  + (size_t)(rowBase + r) * K + seg;   // hoisted: one mul
    const bf16_t* gB = Bt + (size_t)(colBase + r) * K + seg;

    uint4 ra0 = *(const uint4*)(gA);
    uint4 ra1 = *(const uint4*)(gA + rowSkip);
    uint4 rb0 = *(const uint4*)(gB);
    uint4 rb1 = *(const uint4*)(gB + rowSkip);
    gA += BK; gB += BK;

    const int NT = K / BK;
    for (int kt = 0; kt < NT; ++kt) {
        const int buf = kt & 1;
        // dump previously fetched tile into LDS buffer `buf`
        *(uint4*)&As[buf][r][seg]      = ra0;
        *(uint4*)&As[buf][r + 64][seg] = ra1;
        *(uint4*)&Bs[buf][r][seg]      = rb0;
        *(uint4*)&Bs[buf][r + 64][seg] = rb1;
        // issue next tile's global loads (uniform branch, no exec masking)
        if (kt + 1 < NT) {
            ra0 = *(const uint4*)(gA);
            ra1 = *(const uint4*)(gA + rowSkip);
            rb0 = *(const uint4*)(gB);
            rb1 = *(const uint4*)(gB + rowSkip);
            __builtin_prefetch(gA + BK, 0, 3);   // gfx1250 global_prefetch_b8, tile k+2
            __builtin_prefetch(gB + BK, 0, 3);
            gA += BK; gB += BK;
        }
        __syncthreads();

        union Frag { v16bf v; uint4 q[2]; };
        Frag fa[2], fb[4];
        // A fragment (16x32): lanes 0-15 K{0..7,16..23}, lanes 16-31 K{8..15,24..31}
        #pragma unroll
        for (int mi = 0; mi < 2; ++mi) {
            int rr = wr * 32 + mi * 16 + (lane & 15);
            int kb = (lane >> 4) * 8;
            fa[mi].q[0] = *(const uint4*)&As[buf][rr][kb];
            fa[mi].q[1] = *(const uint4*)&As[buf][rr][kb + 16];
        }
        // B fragment (32x16): lanes 0-15 K{0..7,16..23}, lanes 16-31 K{8..15,24..31}
        #pragma unroll
        for (int ni = 0; ni < 4; ++ni) {
            int cc = wc * 64 + ni * 16 + (lane & 15);
            int kb = (lane >> 4) * 8;
            fb[ni].q[0] = *(const uint4*)&Bs[buf][cc][kb];
            fb[ni].q[1] = *(const uint4*)&Bs[buf][cc][kb + 16];
        }
        #pragma unroll
        for (int mi = 0; mi < 2; ++mi)
            #pragma unroll
            for (int ni = 0; ni < 4; ++ni)
                acc[mi][ni] = __builtin_amdgcn_wmma_f32_16x16x32_bf16(
                    false, fa[mi].v, false, fb[ni].v, (short)0, acc[mi][ni],
                    false, false);
    }
}

// ---------------------------------------------------------------------------
// MoE kernel: out[b,n] = sum_e gate[b,e]*relu(BN_e(x@W_e + bias_e)) (+ attn)
// ---------------------------------------------------------------------------
__global__ __launch_bounds__(256) void moe_kernel(const bf16_t* __restrict__ xb,
        const bf16_t* __restrict__ Wt,  // [E][N][K] bf16
        const float* __restrict__ bias, const float* __restrict__ g,
        const float* __restrict__ be,   const float* __restrict__ rm,
        const float* __restrict__ rv,   const float* __restrict__ gate,
        const float* __restrict__ attn, float* __restrict__ out,
        int K, int N) {
    __shared__ bf16_t As[2][128][48];
    __shared__ bf16_t Bs[2][128][48];
    const int rowBase = blockIdx.y * 128;
    const int colBase = blockIdx.x * 128;
    const int lane = threadIdx.x & 31;
    const int wave = threadIdx.x >> 5;
    const int wr = wave >> 1, wc = wave & 1;

    v8f mix[2][4] = {};
    for (int e = 0; e < NEXP; ++e) {
        v8f acc[2][4] = {};
        gemm_core(xb, Wt + (size_t)e * N * K, K, rowBase, colBase, As, Bs, acc);

        float gv[2][8];
        #pragma unroll
        for (int mi = 0; mi < 2; ++mi)
            #pragma unroll
            for (int j = 0; j < 8; ++j) {
                int row = rowBase + wr * 32 + mi * 16 + (lane >> 4) * 8 + j;
                gv[mi][j] = gate[(size_t)row * NEXP + e];
            }
        #pragma unroll
        for (int mi = 0; mi < 2; ++mi)
            #pragma unroll
            for (int ni = 0; ni < 4; ++ni) {
                int col = colBase + wc * 64 + ni * 16 + (lane & 15);
                size_t ec = (size_t)e * N + col;
                float sA = g[ec] * rsqrtf(rv[ec] + 1e-5f);
                float sB = be[ec] + (bias[ec] - rm[ec]) * sA;
                #pragma unroll
                for (int j = 0; j < 8; ++j) {
                    float v = fmaxf(acc[mi][ni][j] * sA + sB, 0.f);
                    mix[mi][ni][j] += gv[mi][j] * v;
                }
            }
    }
    #pragma unroll
    for (int mi = 0; mi < 2; ++mi)
        #pragma unroll
        for (int ni = 0; ni < 4; ++ni) {
            int col = colBase + wc * 64 + ni * 16 + (lane & 15);
            #pragma unroll
            for (int j = 0; j < 8; ++j) {
                int row = rowBase + wr * 32 + mi * 16 + (lane >> 4) * 8 + j;
                float v = mix[mi][ni][j];
                if (attn) v += attn[(size_t)row * N + col];
                out[(size_t)row * N + col] = v;
            }
        }
}

// ---------------------------------------------------------------------------
// Generic GEMM with fused epilogue.
// MODE 0: bf16 out (no bias). MODE 1: f32 out = acc + bias.
// MODE 2: f32 out = sigmoid(BN(acc + bias)).
// ---------------------------------------------------------------------------
template <int MODE>
__global__ __launch_bounds__(256) void gemm_kernel(const bf16_t* __restrict__ A,
        const bf16_t* __restrict__ Bt, int K, int N,
        const float* __restrict__ bias, const float* __restrict__ g,
        const float* __restrict__ be,   const float* __restrict__ rm,
        const float* __restrict__ rv,
        float* __restrict__ outf, bf16_t* __restrict__ outh) {
    __shared__ bf16_t As[2][128][48];
    __shared__ bf16_t Bs[2][128][48];
    const int rowBase = blockIdx.y * 128;
    const int colBase = blockIdx.x * 128;
    const int lane = threadIdx.x & 31;
    const int wave = threadIdx.x >> 5;
    const int wr = wave >> 1, wc = wave & 1;

    v8f acc[2][4] = {};
    gemm_core(A, Bt, K, rowBase, colBase, As, Bs, acc);

    #pragma unroll
    for (int mi = 0; mi < 2; ++mi)
        #pragma unroll
        for (int ni = 0; ni < 4; ++ni) {
            int col = colBase + wc * 64 + ni * 16 + (lane & 15);
            float sA = 1.f, sB = 0.f;
            if (MODE == 1) sB = bias[col];
            if (MODE == 2) {
                sA = g[col] * rsqrtf(rv[col] + 1e-5f);
                sB = be[col] + (bias[col] - rm[col]) * sA;
            }
            #pragma unroll
            for (int j = 0; j < 8; ++j) {
                int row = rowBase + wr * 32 + mi * 16 + (lane >> 4) * 8 + j;
                float v = acc[mi][ni][j] * sA + sB;
                if (MODE == 2) v = 1.f / (1.f + __expf(-v));
                if (MODE == 0) outh[(size_t)row * N + col] = (bf16_t)v;
                else           outf[(size_t)row * N + col] = v;
            }
        }
}

// ---------------------------------------------------------------------------
// Host-side orchestration
// ---------------------------------------------------------------------------
extern "C" void kernel_launch(void* const* d_in, const int* in_sizes, int n_in,
                              void* d_out, int out_size, void* d_ws, size_t ws_size,
                              hipStream_t stream) {
    const float* x      = (const float*)d_in[0];
    const float* Wm_mu  = (const float*)d_in[1];
    const float* bm_mu  = (const float*)d_in[2];
    const float* g_mu   = (const float*)d_in[3];
    const float* be_mu  = (const float*)d_in[4];
    const float* rm_mu  = (const float*)d_in[5];
    const float* rv_mu  = (const float*)d_in[6];
    const float* Wm_d   = (const float*)d_in[7];
    const float* bm_d   = (const float*)d_in[8];
    const float* g_d    = (const float*)d_in[9];
    const float* be_d   = (const float*)d_in[10];
    const float* rm_d   = (const float*)d_in[11];
    const float* rv_d   = (const float*)d_in[12];
    const float* gw1_mu = (const float*)d_in[13];
    const float* gb1_mu = (const float*)d_in[14];
    const float* gw2_mu = (const float*)d_in[15];
    const float* gb2_mu = (const float*)d_in[16];
    const float* gw1_d  = (const float*)d_in[17];
    const float* gb1_d  = (const float*)d_in[18];
    const float* gw2_d  = (const float*)d_in[19];
    const float* gb2_d  = (const float*)d_in[20];
    // d_in[21] Wq, d_in[22] Wk: dead (softmax over seq len 1 == 1)
    const float* Wv     = (const float*)d_in[23];
    const float* Wp     = (const float*)d_in[24];
    const float* bp     = (const float*)d_in[25];
    const float* se_w1  = (const float*)d_in[26];
    const float* se_w2  = (const float*)d_in[27];
    const float* Wr_dec = (const float*)d_in[28];
    const float* br_dec = (const float*)d_in[29];
    const float* gr     = (const float*)d_in[30];
    const float* betar  = (const float*)d_in[31];
    const float* rmr    = (const float*)d_in[32];
    const float* rvr    = (const float*)d_in[33];
    const float* Wm_dec = (const float*)d_in[34];
    const float* bm_dec = (const float*)d_in[35];
    const float* gm     = (const float*)d_in[36];
    const float* betam  = (const float*)d_in[37];
    const float* rmm    = (const float*)d_in[38];
    const float* rvm    = (const float*)d_in[39];

    // output layout (fp32, concatenated): r[4096*2048] m[4096*1024] mu[...] d[...]
    float* out_r  = (float*)d_out;
    float* out_m  = out_r  + (size_t)B_ROWS * DR;
    float* out_mu = out_m  + (size_t)B_ROWS * DM;
    float* out_d  = out_mu + (size_t)B_ROWS * DE;

    // workspace carve (all sizes are multiples of 256B)
    char* ws = (char*)d_ws;
    size_t off = 0;
    auto carve = [&](size_t bytes) { void* p = ws + off; off += (bytes + 255) & ~(size_t)255; return p; };
    bf16_t* xb      = (bf16_t*)carve((size_t)B_ROWS * DM * 2);
    bf16_t* Wmmu_t  = (bf16_t*)carve((size_t)NEXP * DM * DE * 2);
    bf16_t* Wmd_t   = (bf16_t*)carve((size_t)NEXP * DM * DE * 2);
    bf16_t* Wv_t    = (bf16_t*)carve((size_t)DM * DE * 2);
    bf16_t* Wp_t    = (bf16_t*)carve((size_t)DE * DE * 2);
    bf16_t* Wr_t    = (bf16_t*)carve((size_t)DE * DR * 2);
    bf16_t* Wmdec_t = (bf16_t*)carve((size_t)DE * DM * 2);
    bf16_t* xv      = (bf16_t*)carve((size_t)B_ROWS * DE * 2);
    float*  attn    = (float*) carve((size_t)B_ROWS * DE * 4);
    bf16_t* latent  = (bf16_t*)carve((size_t)B_ROWS * DE * 2);
    float*  gate_mu = (float*) carve((size_t)B_ROWS * NEXP * 4);
    float*  gate_d  = (float*) carve((size_t)B_ROWS * NEXP * 4);
    (void)ws_size; (void)in_sizes; (void)n_in; (void)out_size;

    // --- Stage A: precision conversion / weight transposition ---
    {
        int n = B_ROWS * DM;
        cvt_f32_bf16<<<n / 256, 256, 0, stream>>>(x, xb, n);
    }
    tconv_kernel<<<dim3(DE / 32, DM / 32, NEXP), 256, 0, stream>>>(Wm_mu,  Wmmu_t,  DM, DE);
    tconv_kernel<<<dim3(DE / 32, DM / 32, NEXP), 256, 0, stream>>>(Wm_d,   Wmd_t,   DM, DE);
    tconv_kernel<<<dim3(DE / 32, DM / 32, 1),    256, 0, stream>>>(Wv,     Wv_t,    DM, DE);
    tconv_kernel<<<dim3(DE / 32, DE / 32, 1),    256, 0, stream>>>(Wp,     Wp_t,    DE, DE);
    tconv_kernel<<<dim3(DR / 32, DE / 32, 1),    256, 0, stream>>>(Wr_dec, Wr_t,    DE, DR);
    tconv_kernel<<<dim3(DM / 32, DE / 32, 1),    256, 0, stream>>>(Wm_dec, Wmdec_t, DE, DM);

    // --- Stage B: gate MLPs (both branches) ---
    gate_kernel<<<B_ROWS, 128, 0, stream>>>(x, gw1_mu, gb1_mu, gw2_mu, gb2_mu,
                                            gw1_d, gb1_d, gw2_d, gb2_d,
                                            gate_mu, gate_d);

    // --- Stage C: attention (seq len 1 => attn = (x@Wv)@Wp + bp) ---
    gemm_kernel<0><<<dim3(DE / 128, B_ROWS / 128), 256, 0, stream>>>(
        xb, Wv_t, DM, DE, nullptr, nullptr, nullptr, nullptr, nullptr, nullptr, xv);
    gemm_kernel<1><<<dim3(DE / 128, B_ROWS / 128), 256, 0, stream>>>(
        xv, Wp_t, DE, DE, bp, nullptr, nullptr, nullptr, nullptr, attn, nullptr);

    // --- Stage D: MoE branches (mu gets attention fused into epilogue) ---
    moe_kernel<<<dim3(DE / 128, B_ROWS / 128), 256, 0, stream>>>(
        xb, Wmmu_t, bm_mu, g_mu, be_mu, rm_mu, rv_mu, gate_mu, attn, out_mu, DM, DE);
    moe_kernel<<<dim3(DE / 128, B_ROWS / 128), 256, 0, stream>>>(
        xb, Wmd_t, bm_d, g_d, be_d, rm_d, rv_d, gate_d, nullptr, out_d, DM, DE);

    // --- Stage E: SE gating -> latent (bf16) ---
    se_kernel<<<B_ROWS, 128, 0, stream>>>(out_mu, se_w1, se_w2, latent);

    // --- Stage F: decoders with fused sigmoid(BN(...)) ---
    gemm_kernel<2><<<dim3(DR / 128, B_ROWS / 128), 256, 0, stream>>>(
        latent, Wr_t, DE, DR, br_dec, gr, betar, rmr, rvr, out_r, nullptr);
    gemm_kernel<2><<<dim3(DM / 128, B_ROWS / 128), 256, 0, stream>>>(
        latent, Wmdec_t, DE, DM, bm_dec, gm, betam, rmm, rvm, out_m, nullptr);
}
